// BahdanauAttention_197568495808
// MI455X (gfx1250) — compile-verified, matrix operand order checked by reading, not run on hardware
//
#include <hip/hip_runtime.h>
#include <hip/hip_bf16.h>

static constexpr int kH = 1024;   // hidden
static constexpr int kS = 2048;   // sequence
static constexpr int kB = 32;     // batch

typedef __attribute__((ext_vector_type(16))) _Float16 v16h;
typedef __attribute__((ext_vector_type(8)))  _Float16 v8h;
typedef __attribute__((ext_vector_type(8)))  float    v8f;

#if __has_builtin(__builtin_amdgcn_tanhf)
#define TANHF(x) __builtin_amdgcn_tanhf(x)
#else
#define TANHF(x) tanhf(x)
#endif

#if __has_builtin(__builtin_amdgcn_s_wait_asynccnt)
#define WAIT_ASYNC0() __builtin_amdgcn_s_wait_asynccnt(0)
#else
#define WAIT_ASYNC0() asm volatile("s_wait_asynccnt 0x0" ::: "memory")
#endif

// ---------------------------------------------------------------------------
// Kernel 0: fp32 -> f16 conversion of Ua_w (1024x1024, k-contiguous rows)
// ---------------------------------------------------------------------------
__global__ void cvt_f16_kernel(const float* __restrict__ src,
                               _Float16* __restrict__ dst, int n) {
    int i = (blockIdx.x * blockDim.x + threadIdx.x) * 8;
    if (i + 8 <= n) {
        const float4* s = (const float4*)(src + i);
        float4 f0 = s[0], f1 = s[1];
        v8h h;
        h[0] = (_Float16)f0.x; h[1] = (_Float16)f0.y;
        h[2] = (_Float16)f0.z; h[3] = (_Float16)f0.w;
        h[4] = (_Float16)f1.x; h[5] = (_Float16)f1.y;
        h[6] = (_Float16)f1.z; h[7] = (_Float16)f1.w;
        *(v8h*)(dst + i) = h;
    }
}

// ---------------------------------------------------------------------------
// Kernel 1: qpb[b,o] = dot(query[b,:], Wa_w[o,:]) + Wa_b[o] + Ua_b[o]
// ---------------------------------------------------------------------------
__global__ void qproj_kernel(const float* __restrict__ query,
                             const float* __restrict__ Wa_w,
                             const float* __restrict__ Wa_b,
                             const float* __restrict__ Ua_b,
                             float* __restrict__ qpb) {
    __shared__ float q[kH];
    const int b = blockIdx.x;
    const int tid = threadIdx.x;
    for (int i = tid; i < kH; i += 256) q[i] = query[b * kH + i];
    __syncthreads();
    const int o = blockIdx.y * 256 + tid;
    const float* wr = Wa_w + (size_t)o * kH;
    float acc = Wa_b[o] + Ua_b[o];
#pragma unroll 4
    for (int k = 0; k < kH; k += 4) {
        float4 wv = *(const float4*)(wr + k);
        acc += wv.x * q[k] + wv.y * q[k + 1] + wv.z * q[k + 2] + wv.w * q[k + 3];
    }
    qpb[b * kH + o] = acc;
}

// ---------------------------------------------------------------------------
// Kernel 2 (WMMA core): fused k_proj GEMM + tanh + Va dot -> scores[B*S]
// One WG = 64 rows of keys (single batch), 4 waves, 16-row stripe per wave.
// Keys rows live f16 in LDS for the whole kernel. Ua f16 tiles are streamed
// through a double-buffered LDS tile with GLOBAL_LOAD_ASYNC_TO_LDS_B128
// (ASYNCcnt-tracked DMA), software-pipelined one stage ahead of the WMMAs.
// ---------------------------------------------------------------------------
__launch_bounds__(128)
__global__ void score_kernel(const float* __restrict__ keys,
                             const _Float16* __restrict__ Ua16,
                             const float* __restrict__ qpb,
                             const float* __restrict__ Va_w,
                             const float* __restrict__ Va_b,
                             float* __restrict__ scores) {
    // strides padded to stay 16B aligned and spread LDS banks
    __shared__ __attribute__((aligned(16))) _Float16 Atile[64][kH + 8];    // keys f16 (129KB)
    __shared__ __attribute__((aligned(16))) _Float16 BtileBuf[2][64][72];  // Ua f16 x2 (18KB)

    const int tid  = threadIdx.x;
    const int lane = tid & 31;
    const int wave = tid >> 5;
    const int rowbase = blockIdx.x * 64;      // global m for row 0 of tile
    const int b = rowbase >> 11;              // one batch per WG

    // ---- issue async DMA for the first Ua tile (stage 0) ----
    auto issue_stage = [&](int st) {
        const int ot = (st >> 4) << 6;        // N-tile base     (st/16)*64
        const int kb = (st & 15) << 6;        // K-chunk base    (st%16)*64
        const int buf = st & 1;
        for (int idx = tid; idx < 64 * 8; idx += 128) {
            const int row = idx >> 3;
            const int c8  = idx & 7;
            unsigned lds = (unsigned)(uintptr_t)&BtileBuf[buf][row][c8 * 8];
            unsigned long long g = (unsigned long long)(uintptr_t)
                (Ua16 + (size_t)(ot + row) * kH + kb + c8 * 8);
            asm volatile("global_load_async_to_lds_b128 %0, %1, off"
                         :: "v"(lds), "v"(g) : "memory");
        }
    };
    issue_stage(0);

    // ---- stage keys tile: 64 x 1024 fp32 -> f16 in LDS (once) ----
    for (int idx = tid; idx < 64 * (kH / 8); idx += 128) {
        const int row = idx >> 7;
        const int c8  = idx & 127;
        const float4* src = (const float4*)(keys + (size_t)(rowbase + row) * kH + c8 * 8);
        float4 f0 = src[0], f1 = src[1];
        v8h h;
        h[0] = (_Float16)f0.x; h[1] = (_Float16)f0.y;
        h[2] = (_Float16)f0.z; h[3] = (_Float16)f0.w;
        h[4] = (_Float16)f1.x; h[5] = (_Float16)f1.y;
        h[6] = (_Float16)f1.z; h[7] = (_Float16)f1.w;
        *(v8h*)&Atile[row][c8 * 8] = h;
    }

    const int mhalf = lane >> 4;              // 0/1: C rows M=r vs M=r+8
    const int l15   = lane & 15;              // N column (B) / M row (A)
    const int arow  = wave * 16 + l15;        // A-operand row for this lane
    const int khalf = mhalf * 8;              // operand K half-offset

    float sacc[8];
#pragma unroll
    for (int r = 0; r < 8; ++r) sacc[r] = 0.f;
    const float vab = Va_b[0];

    for (int ot_idx = 0; ot_idx < 16; ++ot_idx) {   // 16 N-tiles of 64 chans
        const int ot = ot_idx << 6;
        v8f acc[4];
#pragma unroll
        for (int i = 0; i < 4; ++i)
#pragma unroll
            for (int e = 0; e < 8; ++e) acc[i][e] = 0.f;

        for (int kbi = 0; kbi < 16; ++kbi) {        // 16 K-chunks of 64
            const int st = (ot_idx << 4) | kbi;
            WAIT_ASYNC0();                          // my DMA for buf[st&1] done
            __syncthreads();                        // everyone's DMA + prev reads done
            if (st + 1 < 256) issue_stage(st + 1);  // prefetch next tile (other buf)

            const _Float16 (*Bt)[72] = BtileBuf[st & 1];
            const int kb = kbi << 6;
#pragma unroll
            for (int kk = 0; kk < 64; kk += 32) {
                const int ka = kb + kk + khalf;
                v8h alo = *(const v8h*)&Atile[arow][ka];
                v8h ahi = *(const v8h*)&Atile[arow][ka + 16];
                v16h a = __builtin_shufflevector(alo, ahi,
                        0, 1, 2, 3, 4, 5, 6, 7, 8, 9, 10, 11, 12, 13, 14, 15);
                v16h bv[4];
#pragma unroll
                for (int nt = 0; nt < 4; ++nt) {
                    const int brow = nt * 16 + l15;
                    const int kbl  = kk + khalf;
                    v8h blo = *(const v8h*)&Bt[brow][kbl];
                    v8h bhi = *(const v8h*)&Bt[brow][kbl + 16];
                    bv[nt] = __builtin_shufflevector(blo, bhi,
                            0, 1, 2, 3, 4, 5, 6, 7, 8, 9, 10, 11, 12, 13, 14, 15);
                }
#pragma unroll
                for (int nt = 0; nt < 4; ++nt)
                    acc[nt] = __builtin_amdgcn_wmma_f32_16x16x32_f16(
                        false, a, false, bv[nt], (short)0, acc[nt], false, false);
            }
        }

        // ---- fused epilogue: tanh + Va-weighted N reduction ----
        float rs[8];
#pragma unroll
        for (int r = 0; r < 8; ++r) rs[r] = 0.f;
#pragma unroll
        for (int nt = 0; nt < 4; ++nt) {
            const int n  = ot + nt * 16 + l15;
            const float qp = qpb[b * kH + n];
            const float va = Va_w[n];
#pragma unroll
            for (int r = 0; r < 8; ++r)
                rs[r] += va * TANHF(acc[nt][r] + qp);
        }
#pragma unroll
        for (int r = 0; r < 8; ++r) {               // reduce over N (16 lanes)
            float v = rs[r];
            v += __shfl_xor(v, 1);
            v += __shfl_xor(v, 2);
            v += __shfl_xor(v, 4);
            v += __shfl_xor(v, 8);
            sacc[r] += v;
        }
    }

    if (l15 == 0) {
#pragma unroll
        for (int r = 0; r < 8; ++r) {
            const int mloc = wave * 16 + mhalf * 8 + r;  // C-matrix row mapping
            scores[rowbase + mloc] = sacc[r] + vab;
        }
    }
}

// ---------------------------------------------------------------------------
// Kernel 3: per-batch softmax over S=2048 -> weights
// ---------------------------------------------------------------------------
__global__ void softmax_kernel(const float* __restrict__ scores,
                               float* __restrict__ weights) {
    __shared__ float red[256];
    const int b = blockIdx.x, tid = threadIdx.x;
    float v[8];
    float mx = -3.4e38f;
#pragma unroll
    for (int i = 0; i < 8; ++i) {
        v[i] = scores[b * kS + i * 256 + tid];
        mx = fmaxf(mx, v[i]);
    }
    red[tid] = mx;
    __syncthreads();
    for (int off = 128; off > 0; off >>= 1) {
        if (tid < off) red[tid] = fmaxf(red[tid], red[tid + off]);
        __syncthreads();
    }
    mx = red[0];
    __syncthreads();
    float s = 0.f;
#pragma unroll
    for (int i = 0; i < 8; ++i) { v[i] = __expf(v[i] - mx); s += v[i]; }
    red[tid] = s;
    __syncthreads();
    for (int off = 128; off > 0; off >>= 1) {
        if (tid < off) red[tid] += red[tid + off];
        __syncthreads();
    }
    const float inv = 1.f / red[0];
#pragma unroll
    for (int i = 0; i < 8; ++i)
        weights[b * kS + i * 256 + tid] = v[i] * inv;
}

// ---------------------------------------------------------------------------
// Kernel 4: context[b,h] = sum_s weights[b,s] * keys[b,s,h]  (coalesced in h)
// ---------------------------------------------------------------------------
__global__ void context_kernel(const float* __restrict__ weights,
                               const float* __restrict__ keys,
                               float* __restrict__ context) {
    const int b = blockIdx.x;
    const int h = blockIdx.y * 256 + threadIdx.x;
    const float* kp = keys + (size_t)b * kS * kH + h;
    const float* wp = weights + b * kS;
    float acc = 0.f;
#pragma unroll 4
    for (int s = 0; s < kS; ++s)
        acc += wp[s] * kp[(size_t)s * kH];
    context[b * kH + h] = acc;
}

// ---------------------------------------------------------------------------
extern "C" void kernel_launch(void* const* d_in, const int* in_sizes, int n_in,
                              void* d_out, int out_size, void* d_ws, size_t ws_size,
                              hipStream_t stream) {
    const float* query = (const float*)d_in[0];
    const float* keys  = (const float*)d_in[1];
    const float* Wa_w  = (const float*)d_in[2];
    const float* Wa_b  = (const float*)d_in[3];
    const float* Ua_w  = (const float*)d_in[4];
    const float* Ua_b  = (const float*)d_in[5];
    const float* Va_w  = (const float*)d_in[6];
    const float* Va_b  = (const float*)d_in[7];

    float* out     = (float*)d_out;
    float* context = out;               // [B,1,H] = 32768 floats
    float* weights = out + kB * kH;     // [B,1,S] = 65536 floats

    char* w = (char*)d_ws;
    _Float16* Ua16 = (_Float16*)w;                                    // 2 MB
    float* qpb     = (float*)(w + (size_t)kH * kH * 2);               // 128 KB
    float* scores  = (float*)(w + (size_t)kH * kH * 2 + kB * kH * 4); // 256 KB

    cvt_f16_kernel<<<(kH * kH) / (256 * 8), 256, 0, stream>>>(Ua_w, Ua16, kH * kH);
    qproj_kernel<<<dim3(kB, kH / 256), 256, 0, stream>>>(query, Wa_w, Wa_b, Ua_b, qpb);
    score_kernel<<<(kB * kS) / 64, 128, 0, stream>>>(keys, Ua16, qpb, Va_w, Va_b, scores);
    softmax_kernel<<<kB, 256, 0, stream>>>(scores, weights);
    context_kernel<<<dim3(kB, kH / 256), 256, 0, stream>>>(weights, keys, context);
}
